// BitwiseCellularAutomata_84104049590737
// MI455X (gfx1250) — matrix operand order repaired and checked by reading, not compiled, not run on hardware
//
#include <hip/hip_runtime.h>
#include <hip/hip_bf16.h>

// Reference collapses analytically: W0 = I  =>  step 1 gives E_new = E ^ (I@E>0) = 0,
// which freezes the whole system (wiring = 0, W stays I, P stays P0) for all later steps.
// Exact result for ANY input:  h_out = h0 & 7,  W_out = identity (per batch).
// => pure streaming-store problem: 268.7 MB of writes, ~11.5 us at 23.3 TB/s.
// Diagonal 1.0 tiles (the only nonzero content of W) are produced by
// v_wmma_f32_16x16x32_f16 computing D = I x I + 0 and stored via the ISA D-layout.

typedef __attribute__((ext_vector_type(16))) _Float16 v16h;
typedef __attribute__((ext_vector_type(8)))  float    v8f;
typedef __attribute__((ext_vector_type(4)))  float    v4f;   // native vector for NT b128 stores

#define BB 16
#define NN 2048

__global__ __launch_bounds__(256) void
ca_h_kernel(const int* __restrict__ stim, float* __restrict__ h_out) {
    int i = blockIdx.x * 256 + threadIdx.x;   // exactly BB*NN threads launched
    h_out[i] = (float)(stim[i] & 7);          // clear energy bit, keep P,G,L
}

// One block per (batch, 16-row slab): zero-fill 16x2048 floats with NT b128 stores
// (skipping the 16x16 diagonal block), then wave 0 emits the identity tile via WMMA.
__global__ __launch_bounds__(256) void
ca_w_kernel(float* __restrict__ W) {
    const int tid  = threadIdx.x;
    const int bidx = blockIdx.x >> 7;        // batch        (NN/16 == 128 slabs/batch)
    const int rblk = blockIdx.x & 127;       // 16-row slab index
    const int r0   = rblk << 4;              // first row / diagonal column base

    float* slab = W + ((size_t)bidx * NN + (size_t)r0) * NN;   // 16 x 2048 floats

    // ---- bulk zeros: 8192 float4 slots, lane-consecutive => 512B per wave store ----
    const int diag4 = rblk << 2;             // float4 column range of diagonal block
    v4f  z  = {};                            // 0,0,0,0
    v4f* p4 = (v4f*)slab;
    #pragma unroll
    for (int i = 0; i < 32; ++i) {
        int slot = i * 256 + tid;            // [0, 8192)
        int col4 = slot & 511;               // float4 column within row
        if (col4 < diag4 || col4 >= diag4 + 4)
            __builtin_nontemporal_store(z, p4 + slot);
    }

    // ---- wave 0: 16x16 identity tile = WMMA(I, I, 0), stored on the diagonal ----
    if (tid < 32) {                          // wave-uniform branch: EXEC all-ones inside
        const int l = tid;
        v16h a, b;
        v8f  c = {};
        #pragma unroll
        for (int j = 0; j < 16; ++j) {
            // A (16x32 f16): row m = l%16; lane l<16 holds K=0..7 (halves 0..7) and
            // K=16..23 (halves 8..15); lane l>=16 holds K=8..15 and 24..31.
            // Identity A[m,m]=1 => lanes 0..7 set half l; lanes 24..31 set half l-24.
            bool aset = (l < 8) ? (j == l) : (l >= 24 && j == (l - 24));
            // B (32x16 f16): column n = l%16; same K layout as A.
            // Identity B[n,n]=1 => lanes 0..7 set half l; lanes 24..31 set half l-24.
            bool bset = (l < 8) ? (j == l) : (l >= 24 && j == (l - 24));
            a[j] = (_Float16)(aset ? 1.0f : 0.0f);
            b[j] = (_Float16)(bset ? 1.0f : 0.0f);
        }
        v8f d = __builtin_amdgcn_wmma_f32_16x16x32_f16(
                    /*neg_a=*/false, a, /*neg_b=*/false, b,
                    /*c_mod=*/(short)0, c, /*reuse_a=*/false, /*reuse_b=*/false);
        // D layout: VGPR r, lanes 0-15 -> row r, lanes 16-31 -> row r+8; col = l&15.
        const int col = l & 15;
        const int rhi = (l >= 16) ? 8 : 0;
        #pragma unroll
        for (int r = 0; r < 8; ++r) {
            int row = r + rhi;
            __builtin_nontemporal_store(d[r], slab + (size_t)row * NN + (r0 + col));
        }
    }
}

extern "C" void kernel_launch(void* const* d_in, const int* in_sizes, int n_in,
                              void* d_out, int out_size, void* d_ws, size_t ws_size,
                              hipStream_t stream) {
    const int* stim = (const int*)d_in[0];   // [16, 2048] int32, values 0..15
    float* out = (float*)d_out;              // [h (32768) | W (16*2048*2048)] floats

    // h = stimuli & 7
    ca_h_kernel<<<(BB * NN) / 256, 256, 0, stream>>>(stim, out);

    // W = identity per batch: 2048 blocks x (16-row slab of one batch)
    ca_w_kernel<<<BB * (NN / 16), 256, 0, stream>>>(out + BB * NN);
}
